// MultiHeadAttention_50483045597697
// MI455X (gfx1250) — matrix operand order repaired and checked by reading, not compiled, not run on hardware
//
#include <hip/hip_runtime.h>

// ---------------------------------------------------------------------------
// MHA forward for MI455X (gfx1250, wave32, WMMA bf16 -> f32 accumulate)
//   B=2, S=2048, D=1024, H=16, Hd=64
//   d_out = [ out (2*2048*1024 f32) | attn (2*16*2048*2048 f32) ]
// ---------------------------------------------------------------------------

#define D_MODEL  1024
#define N_HEAD   16
#define HEAD_DIM 64
#define BATCH    2
#define SEQ      2048
#define M_ROWS   (BATCH * SEQ)      // 4096
#define BH       (BATCH * N_HEAD)   // 32

typedef __bf16 bf16_t;
typedef __attribute__((ext_vector_type(16))) __bf16 v16bf;
typedef __attribute__((ext_vector_type(8)))  __bf16 v8bf;
typedef __attribute__((ext_vector_type(8)))  float  v8f;

union ABFrag { v16bf v; v8bf h[2]; };

// A fragment (16x32 bf16, M x K):
//   lane L: row M = L%16; lanes 0-15 hold K = k0+[0..7] and k0+16+[0..7];
//   lanes 16-31 hold K = k0+8+[0..7] and k0+24+[0..7].
__device__ __forceinline__ v16bf load_a_frag_bf16(const bf16_t* __restrict__ rowptr,
                                                  int k0, int lhalf) {
  ABFrag u;
  u.h[0] = *(const v8bf*)(rowptr + k0 + lhalf * 8);
  u.h[1] = *(const v8bf*)(rowptr + k0 + 16 + lhalf * 8);
  return u.v;
}

// B fragment (32x16 bf16, K x N): lane L: column N = L%16;
//   lanes 0-15 hold K = k0+[0..7] and k0+16+[0..7]; lanes 16-31 hold K = k0+8+[0..7] and k0+24+[0..7].
//   colptr points at the K-contiguous data for this lane's column.
__device__ __forceinline__ v16bf load_b_frag_bf16(const bf16_t* __restrict__ colptr,
                                                  int k0, int lhalf) {
  ABFrag u;
  const bf16_t* p = colptr + k0 + lhalf * 8;
  u.h[0] = *(const v8bf*)(p);
  u.h[1] = *(const v8bf*)(p + 16);
  return u.v;
}

// A fragment from f32 source (convert to bf16 on the fly) — used for attn @ V.
__device__ __forceinline__ v16bf load_a_frag_f32(const float* __restrict__ rowptr,
                                                 int k0, int lhalf) {
  const float* p0 = rowptr + k0 + lhalf * 8;
  const float* p1 = rowptr + k0 + 16 + lhalf * 8;
  ABFrag u;
#pragma unroll
  for (int i = 0; i < 8; ++i) {
    u.v[i]     = (__bf16)p0[i];
    u.v[i + 8] = (__bf16)p1[i];
  }
  return u.v;
}

__device__ __forceinline__ v8f wmma_bf16(v16bf a, v16bf b, v8f c) {
  // D = A x B + C, f32 accumulate; emits v_wmma_f32_16x16x32_bf16
  return __builtin_amdgcn_wmma_f32_16x16x32_bf16(
      /*neg_a=*/false, a, /*neg_b=*/false, b,
      /*c_mod=*/(short)0, c, /*reuse_a=*/false, /*reuse_b=*/false);
}

// ---------------------------------------------------------------------------
// Elementwise f32 -> bf16 conversion
// ---------------------------------------------------------------------------
__global__ void convert_f32_to_bf16(const float* __restrict__ src,
                                    bf16_t* __restrict__ dst, int n) {
  int i = blockIdx.x * blockDim.x + threadIdx.x;
  int stride = gridDim.x * blockDim.x;
  for (; i < n; i += stride) dst[i] = (__bf16)src[i];
}

// ---------------------------------------------------------------------------
// Projection GEMM: Y[m,n] = sum_k X[m,k] * W[n,k] + bias[n], scaled.
// One wave per 64x32 tile (4x2 WMMA fragments), K = 1024.
// __launch_bounds__(32,1): single-wave block, allow full VGPR budget (no spills)
// OUT_MODE 0: store bf16 per-head layout [B,H,S,Hd]   (Q, K)
// OUT_MODE 1: store bf16 transposed     [B,H,Hd,S]    (V^T), vectorized store
// ---------------------------------------------------------------------------
template <int OUT_MODE>
__global__ void __launch_bounds__(32, 1)
proj_kernel(const bf16_t* __restrict__ X,   // [4096,1024]
            const bf16_t* __restrict__ W,   // [1024,1024]
            const float*  __restrict__ bias,// [1024] f32
            bf16_t* __restrict__ out,
            float scale) {
  constexpr int TM = 4, TN = 2;
  const int lane  = threadIdx.x & 31;
  const int lidx  = lane & 15;
  const int lhalf = lane >> 4;
  const int m0 = blockIdx.x * (TM * 16);
  const int n0 = blockIdx.y * (TN * 16);

  const bf16_t* arow[TM];
  const bf16_t* bcol[TN];
#pragma unroll
  for (int r = 0; r < TM; ++r)
    arow[r] = X + (size_t)(m0 + r * 16 + lidx) * D_MODEL;
#pragma unroll
  for (int c = 0; c < TN; ++c)
    bcol[c] = W + (size_t)(n0 + c * 16 + lidx) * D_MODEL;

  v8f acc[TM][TN] = {};
#pragma unroll 2
  for (int k0 = 0; k0 < D_MODEL; k0 += 32) {
    v16bf a[TM], b[TN];
#pragma unroll
    for (int r = 0; r < TM; ++r) a[r] = load_a_frag_bf16(arow[r], k0, lhalf);
#pragma unroll
    for (int c = 0; c < TN; ++c) b[c] = load_b_frag_bf16(bcol[c], k0, lhalf);
#pragma unroll
    for (int r = 0; r < TM; ++r)
#pragma unroll
      for (int c = 0; c < TN; ++c) acc[r][c] = wmma_bf16(a[r], b[c], acc[r][c]);
  }

#pragma unroll
  for (int r = 0; r < TM; ++r) {
#pragma unroll
    for (int c = 0; c < TN; ++c) {
      const int n  = n0 + c * 16 + lidx;     // D/C layout: N = lane%16
      const float bv = bias[n];
      const int h = n >> 6, d = n & 63;
      if (OUT_MODE == 0) {
#pragma unroll
        for (int v = 0; v < 8; ++v) {
          const int m  = m0 + r * 16 + lhalf * 8 + v;  // M = vgpr + 8*(lane/16)
          const int bi = m >> 11;                      // / SEQ
          const int s  = m & (SEQ - 1);
          const float val = (acc[r][c][v] + bv) * scale;
          out[(((size_t)bi * N_HEAD + h) * SEQ + s) * HEAD_DIM + d] = (__bf16)val;
        }
      } else {
        // lane's 8 accum elements are 8 consecutive sequence positions -> b128
        const int mbase = m0 + r * 16 + lhalf * 8;
        const int bi = mbase >> 11;
        const int s0 = mbase & (SEQ - 1);
        v8bf pack;
#pragma unroll
        for (int v = 0; v < 8; ++v)
          pack[v] = (__bf16)((acc[r][c][v] + bv) * scale);
        *(v8bf*)(out + (((size_t)bi * N_HEAD + h) * HEAD_DIM + d) * SEQ + s0) = pack;
      }
    }
  }
}

// ---------------------------------------------------------------------------
// Scores GEMM: raw[q,kk] = sum_d Q[q,d] * K[kk,d]   (scale folded into Q)
// Per (b,h) slice; one wave per 64x64 tile (4x4 fragments), K-dim = 64.
// Writes raw f32 scores into the attn output region.
// ---------------------------------------------------------------------------
__global__ void __launch_bounds__(32, 1)
scores_kernel(const bf16_t* __restrict__ Q,   // [BH,S,64]
              const bf16_t* __restrict__ Km,  // [BH,S,64]
              float* __restrict__ attn) {     // [BH,S,S]
  constexpr int TM = 4, TN = 4;
  const int lane  = threadIdx.x & 31;
  const int lidx  = lane & 15;
  const int lhalf = lane >> 4;
  const int bh = blockIdx.z;
  const int q0 = blockIdx.x * (TM * 16);
  const int n0 = blockIdx.y * (TN * 16);

  const bf16_t* qb = Q  + (size_t)bh * SEQ * HEAD_DIM;
  const bf16_t* kb = Km + (size_t)bh * SEQ * HEAD_DIM;
  const bf16_t* arow[TM];
  const bf16_t* bcol[TN];
#pragma unroll
  for (int r = 0; r < TM; ++r)
    arow[r] = qb + (size_t)(q0 + r * 16 + lidx) * HEAD_DIM;
#pragma unroll
  for (int c = 0; c < TN; ++c)
    bcol[c] = kb + (size_t)(n0 + c * 16 + lidx) * HEAD_DIM;

  v8f acc[TM][TN] = {};
#pragma unroll
  for (int k0 = 0; k0 < HEAD_DIM; k0 += 32) {
    v16bf a[TM], b[TN];
#pragma unroll
    for (int r = 0; r < TM; ++r) a[r] = load_a_frag_bf16(arow[r], k0, lhalf);
#pragma unroll
    for (int c = 0; c < TN; ++c) b[c] = load_b_frag_bf16(bcol[c], k0, lhalf);
#pragma unroll
    for (int r = 0; r < TM; ++r)
#pragma unroll
      for (int c = 0; c < TN; ++c) acc[r][c] = wmma_bf16(a[r], b[c], acc[r][c]);
  }

  float* obase = attn + (size_t)bh * SEQ * SEQ;
#pragma unroll
  for (int r = 0; r < TM; ++r)
#pragma unroll
    for (int c = 0; c < TN; ++c)
#pragma unroll
      for (int v = 0; v < 8; ++v) {
        const int q = q0 + r * 16 + lhalf * 8 + v;
        const int n = n0 + c * 16 + lidx;
        obase[(size_t)q * SEQ + n] = acc[r][c][v];
      }
}

// ---------------------------------------------------------------------------
// In-place row softmax over attn rows (row length = SEQ = 2048).
// One 256-thread block per row, 8 strided elements per thread.
// ---------------------------------------------------------------------------
__global__ void softmax_kernel(float* __restrict__ attn) {
  const int tid = threadIdx.x;
  float* row = attn + (size_t)blockIdx.x * SEQ;

  float vals[8];
  float m = -3.402823466e38f;
#pragma unroll
  for (int i = 0; i < 8; ++i) {
    vals[i] = row[tid + 256 * i];
    m = fmaxf(m, vals[i]);
  }

  __shared__ float red[256];
  red[tid] = m;
  __syncthreads();
  for (int s = 128; s > 0; s >>= 1) {
    if (tid < s) red[tid] = fmaxf(red[tid], red[tid + s]);
    __syncthreads();
  }
  m = red[0];
  __syncthreads();

  float sum = 0.0f;
#pragma unroll
  for (int i = 0; i < 8; ++i) {
    vals[i] = __expf(vals[i] - m);
    sum += vals[i];
  }
  red[tid] = sum;
  __syncthreads();
  for (int s = 128; s > 0; s >>= 1) {
    if (tid < s) red[tid] += red[tid + s];
    __syncthreads();
  }
  const float inv = 1.0f / red[0];

#pragma unroll
  for (int i = 0; i < 8; ++i) row[tid + 256 * i] = vals[i] * inv;
}

// ---------------------------------------------------------------------------
// Context GEMM: ctx[q,d] = sum_k attn[q,k] * V[k,d], with V^T [BH,64,S] bf16,
// attn f32 converted to bf16 fragments on the fly. K-dim = SEQ = 2048.
// One wave per 32x64 tile (2x4 fragments); ctx stored bf16 [B,S,D] row-major.
// ---------------------------------------------------------------------------
__global__ void __launch_bounds__(32, 1)
ctx_kernel(const float*  __restrict__ attn, // [BH,S,S] f32
           const bf16_t* __restrict__ vt,   // [BH,64,S] bf16
           bf16_t* __restrict__ ctx) {      // [4096,1024] bf16
  constexpr int TM = 2, TN = 4;
  const int lane  = threadIdx.x & 31;
  const int lidx  = lane & 15;
  const int lhalf = lane >> 4;
  const int bh = blockIdx.z;
  const int q0 = blockIdx.x * (TM * 16);
  const int n0 = 0;                      // TN*16 == HEAD_DIM covers all columns

  const float*  ab = attn + (size_t)bh * SEQ * SEQ;
  const bf16_t* vb = vt   + (size_t)bh * HEAD_DIM * SEQ;
  const float*  arow[TM];
  const bf16_t* bcol[TN];
#pragma unroll
  for (int r = 0; r < TM; ++r)
    arow[r] = ab + (size_t)(q0 + r * 16 + lidx) * SEQ;
#pragma unroll
  for (int c = 0; c < TN; ++c)
    bcol[c] = vb + (size_t)(n0 + c * 16 + lidx) * SEQ;

  v8f acc[TM][TN] = {};
#pragma unroll 2
  for (int k0 = 0; k0 < SEQ; k0 += 32) {
    v16bf a[TM], b[TN];
#pragma unroll
    for (int r = 0; r < TM; ++r) a[r] = load_a_frag_f32(arow[r], k0, lhalf);
#pragma unroll
    for (int c = 0; c < TN; ++c) b[c] = load_b_frag_bf16(bcol[c], k0, lhalf);
#pragma unroll
    for (int r = 0; r < TM; ++r)
#pragma unroll
      for (int c = 0; c < TN; ++c) acc[r][c] = wmma_bf16(a[r], b[c], acc[r][c]);
  }

  const int b_i = bh >> 4;
  const int h   = bh & 15;
#pragma unroll
  for (int r = 0; r < TM; ++r)
#pragma unroll
    for (int c = 0; c < TN; ++c) {
      const int d = n0 + c * 16 + lidx;
#pragma unroll
      for (int v = 0; v < 8; ++v) {
        const int q = q0 + r * 16 + lhalf * 8 + v;
        ctx[(size_t)(b_i * SEQ + q) * D_MODEL + h * HEAD_DIM + d] =
            (__bf16)acc[r][c][v];
      }
    }
}

// ---------------------------------------------------------------------------
// Output projection: out[m,n] = sum_k ctx[m,k]*Wo[n,k] + bo[n]  (f32 out)
// One wave per 64x32 tile (4x2 fragments), K = 1024.
// ---------------------------------------------------------------------------
__global__ void __launch_bounds__(32, 1)
out_proj_kernel(const bf16_t* __restrict__ X,   // [4096,1024]
                const bf16_t* __restrict__ W,   // [1024,1024]
                const float*  __restrict__ bias,
                float* __restrict__ out) {      // [4096,1024]
  constexpr int TM = 4, TN = 2;
  const int lane  = threadIdx.x & 31;
  const int lidx  = lane & 15;
  const int lhalf = lane >> 4;
  const int m0 = blockIdx.x * (TM * 16);
  const int n0 = blockIdx.y * (TN * 16);

  const bf16_t* arow[TM];
  const bf16_t* bcol[TN];
#pragma unroll
  for (int r = 0; r < TM; ++r)
    arow[r] = X + (size_t)(m0 + r * 16 + lidx) * D_MODEL;
#pragma unroll
  for (int c = 0; c < TN; ++c)
    bcol[c] = W + (size_t)(n0 + c * 16 + lidx) * D_MODEL;

  v8f acc[TM][TN] = {};
#pragma unroll 2
  for (int k0 = 0; k0 < D_MODEL; k0 += 32) {
    v16bf a[TM], b[TN];
#pragma unroll
    for (int r = 0; r < TM; ++r) a[r] = load_a_frag_bf16(arow[r], k0, lhalf);
#pragma unroll
    for (int c = 0; c < TN; ++c) b[c] = load_b_frag_bf16(bcol[c], k0, lhalf);
#pragma unroll
    for (int r = 0; r < TM; ++r)
#pragma unroll
      for (int c = 0; c < TN; ++c) acc[r][c] = wmma_bf16(a[r], b[c], acc[r][c]);
  }

#pragma unroll
  for (int r = 0; r < TM; ++r)
#pragma unroll
    for (int c = 0; c < TN; ++c) {
      const int n  = n0 + c * 16 + lidx;
      const float bv = bias[n];
#pragma unroll
      for (int v = 0; v < 8; ++v) {
        const int m = m0 + r * 16 + lhalf * 8 + v;
        out[(size_t)m * D_MODEL + n] = acc[r][c][v] + bv;
      }
    }
}

// ---------------------------------------------------------------------------
// Host launch
// ---------------------------------------------------------------------------
extern "C" void kernel_launch(void* const* d_in, const int* in_sizes, int n_in,
                              void* d_out, int out_size, void* d_ws, size_t ws_size,
                              hipStream_t stream) {
  const float* query = (const float*)d_in[0];
  const float* key_i = (const float*)d_in[1];
  const float* value = (const float*)d_in[2];
  const float* wq_w  = (const float*)d_in[3];
  const float* wq_b  = (const float*)d_in[4];
  const float* wk_w  = (const float*)d_in[5];
  const float* wk_b  = (const float*)d_in[6];
  const float* wv_w  = (const float*)d_in[7];
  const float* wv_b  = (const float*)d_in[8];
  const float* wo_w  = (const float*)d_in[9];
  const float* wo_b  = (const float*)d_in[10];

  float* out  = (float*)d_out;                         // [B,S,D]
  float* attn = out + (size_t)M_ROWS * D_MODEL;        // [B,H,S,S]

  // Workspace carve-out (~64 MB total), 256B aligned slices.
  char* ws = (char*)d_ws;
  size_t off = 0;
  auto carve = [&](size_t bytes) -> void* {
    void* p = ws + off;
    off += (bytes + 255) & ~(size_t)255;
    return p;
  };
  const size_t n_act = (size_t)M_ROWS * D_MODEL;   // 4,194,304
  const size_t n_wgt = (size_t)D_MODEL * D_MODEL;  // 1,048,576
  bf16_t* xq  = (bf16_t*)carve(n_act * 2);
  bf16_t* xk  = (bf16_t*)carve(n_act * 2);
  bf16_t* xv  = (bf16_t*)carve(n_act * 2);
  bf16_t* wqb = (bf16_t*)carve(n_wgt * 2);
  bf16_t* wkb = (bf16_t*)carve(n_wgt * 2);
  bf16_t* wvb = (bf16_t*)carve(n_wgt * 2);
  bf16_t* wob = (bf16_t*)carve(n_wgt * 2);
  bf16_t* qh  = (bf16_t*)carve(n_act * 2);   // [B,H,S,64]
  bf16_t* kh  = (bf16_t*)carve(n_act * 2);   // [B,H,S,64]
  bf16_t* vth = (bf16_t*)carve(n_act * 2);   // [B,H,64,S]
  bf16_t* ctx = (bf16_t*)carve(n_act * 2);   // [B*S, D]
  (void)ws_size; (void)in_sizes; (void)n_in; (void)out_size;

  const dim3 cblk(256);
  convert_f32_to_bf16<<<dim3(2048), cblk, 0, stream>>>(query, xq, (int)n_act);
  convert_f32_to_bf16<<<dim3(2048), cblk, 0, stream>>>(key_i, xk, (int)n_act);
  convert_f32_to_bf16<<<dim3(2048), cblk, 0, stream>>>(value, xv, (int)n_act);
  convert_f32_to_bf16<<<dim3(512),  cblk, 0, stream>>>(wq_w, wqb, (int)n_wgt);
  convert_f32_to_bf16<<<dim3(512),  cblk, 0, stream>>>(wk_w, wkb, (int)n_wgt);
  convert_f32_to_bf16<<<dim3(512),  cblk, 0, stream>>>(wv_w, wvb, (int)n_wgt);
  convert_f32_to_bf16<<<dim3(512),  cblk, 0, stream>>>(wo_w, wob, (int)n_wgt);

  const dim3 wave(32);
  const float inv_sqrt_hd = 0.125f;  // 1/sqrt(64), folded into Q
  // 64x32 tiles: grid (4096/64, 1024/32) = (64, 32)
  proj_kernel<0><<<dim3(M_ROWS / 64, D_MODEL / 32), wave, 0, stream>>>(
      xq, wqb, wq_b, qh, inv_sqrt_hd);
  proj_kernel<0><<<dim3(M_ROWS / 64, D_MODEL / 32), wave, 0, stream>>>(
      xk, wkb, wk_b, kh, 1.0f);
  proj_kernel<1><<<dim3(M_ROWS / 64, D_MODEL / 32), wave, 0, stream>>>(
      xv, wvb, wv_b, vth, 1.0f);

  // 64x64 tiles: grid (2048/64, 2048/64, 32)
  scores_kernel<<<dim3(SEQ / 64, SEQ / 64, BH), wave, 0, stream>>>(qh, kh, attn);
  softmax_kernel<<<dim3(BH * SEQ), dim3(256), 0, stream>>>(attn);
  // 32x64 tiles: grid (2048/32, 1, 32)
  ctx_kernel<<<dim3(SEQ / 32, 1, BH), wave, 0, stream>>>(attn, vth, ctx);
  out_proj_kernel<<<dim3(M_ROWS / 64, D_MODEL / 32), wave, 0, stream>>>(
      ctx, wob, wo_b, out);
}
